// MultiHeadAttention_60748017435377
// MI455X (gfx1250) — matrix operand order repaired and checked by reading, not compiled, not run on hardware
//
#include <hip/hip_runtime.h>

// ---------------------------------------------------------------------------
// MI455X (gfx1250) multi-head attention, bf16 WMMA pipeline with f32 accum.
// Shapes: B=2, S=2048, D=1024, H=16, DQK=DV=64, DOUT=1024.
// All matrix math on v_wmma_f32_16x16x32_bf16. B-side tiles are staged
// TRANSPOSED in LDS so every fragment load is 2x ds_load_b128; C tiles are
// staged in LDS and written back as 64B global_store_b128.
// ---------------------------------------------------------------------------

typedef __attribute__((ext_vector_type(16))) __bf16 bf16x16;
typedef __attribute__((ext_vector_type(8)))  float  floatx8;

namespace cfg {
constexpr int B    = 2;
constexpr int S    = 2048;
constexpr int D    = 1024;
constexpr int H    = 16;
constexpr int DQK  = 64;
constexpr int DV   = 64;
constexpr int DOUT = 1024;
constexpr int HDV  = H * DV;     // 1024
constexpr float SCALE = 0.02209708691207961f;  // 1/sqrt(S) per reference
}

__device__ __forceinline__ void lds_fence() {
  // Per-wave LDS RAW ordering + compiler barrier around LDS round trips.
  asm volatile("s_wait_dscnt 0x0" ::: "memory");
}

// --- WMMA fragment loaders (layouts per CDNA5 ISA §7.12.2) ------

// A fragment: 16x32 bf16 (MxK), LDS row-major. Contiguous run of 16 ->
// the compiler emits 2x ds_load_b128 per fragment.
__device__ __forceinline__ bf16x16 load_a_frag(const __bf16* base, int ld, int lane) {
  const int m = lane & 15, g = lane >> 4;
  bf16x16 v;
#pragma unroll
  for (int i = 0; i < 16; ++i) {
    const int k = g * 16 + i;
    v[i] = base[m * ld + k];
  }
  return v;
}

// B fragment from an N-major (transposed) LDS buffer: elem(k,n)=base[n*ld+k].
// Lane reads 16 contiguous bf16 -> 2x ds_load_b128.
__device__ __forceinline__ bf16x16 load_b_fragT(const __bf16* base, int ld, int lane) {
  const int n = lane & 15, g = lane >> 4;
  bf16x16 v;
#pragma unroll
  for (int i = 0; i < 16; ++i) v[i] = base[n * ld + g * 16 + i];
  return v;
}

__device__ __forceinline__ floatx8 wmma_bf16(bf16x16 a, bf16x16 b, floatx8 c) {
  // 8 args: (neg_a, A, neg_b, B, c_mod, C, reuse_a, reuse_b)
  return __builtin_amdgcn_wmma_f32_16x16x32_bf16(false, a, false, b,
                                                 (short)0, c, false, false);
}

// Stage a 32(k) x 64(n) row-major global tile into LDS TRANSPOSED as
// l[n][k] (64x32). 128 threads: thread = (kpair, ngroup). Two b128 global
// loads -> integer lo/hi merge -> 8 packed ds_store_b32. No scalar traffic.
__device__ __forceinline__ void stage_bT(const __bf16* __restrict__ g, size_t ld,
                                         __bf16* __restrict__ l, int t) {
  const int kp = t >> 3;          // 0..15 : k row pair (2*kp, 2*kp+1)
  const int n8 = (t & 7) * 8;     // column group of 8
  const uint4 a = *(const uint4*)&g[(size_t)(2 * kp) * ld + n8];
  const uint4 b = *(const uint4*)&g[(size_t)(2 * kp + 1) * ld + n8];
  uint* l32 = (uint*)l;           // l viewed as [64][16] uints
  const uint* ap = (const uint*)&a;
  const uint* bp = (const uint*)&b;
#pragma unroll
  for (int i = 0; i < 4; ++i) {
    l32[(n8 + 2 * i)     * 16 + kp] = (ap[i] & 0xffffu) | (bp[i] << 16);
    l32[(n8 + 2 * i + 1) * 16 + kp] = (ap[i] >> 16)     | (bp[i] & 0xffff0000u);
  }
}

// --- f32 -> bf16 conversion ------------------------------------------------

__global__ __launch_bounds__(256) void cvt_f32_bf16_kernel(
    const float* __restrict__ in, __bf16* __restrict__ out, int n) {
  int i = blockIdx.x * blockDim.x + threadIdx.x;
  if (i < n) out[i] = (__bf16)in[i];
}

// --- Per-head projection GEMM: (S x D) @ (D x 64) for each (b,h) -----------
// 128 threads (4 waves). Block tile 64x64; wave w: rows [w*16,w*16+16).

__global__ __launch_bounds__(128) void proj_gemm_kernel(
    const __bf16* __restrict__ X,   // (B, S, D) bf16
    const __bf16* __restrict__ W,   // (H, D, 64) bf16
    __bf16* __restrict__ Out) {     // (B, H, S, 64) bf16
  using namespace cfg;
  __shared__ __align__(16) char smem_raw[8192];
  __bf16* As = (__bf16*)smem_raw;            // 64x32 row-major   (4KB)
  __bf16* Wl = (__bf16*)(smem_raw + 4096);   // 64(n) x 32(k) T   (4KB)

  const int z = blockIdx.z;             // b*H + h
  const int b = z / H, h = z % H;
  const __bf16* A  = X + (size_t)b * S * D;
  const __bf16* Wh = W + (size_t)h * D * DQK;
  __bf16*       C  = Out + (size_t)z * S * DQK;

  const int row0 = blockIdx.x * 64;
  const int t    = threadIdx.x;
  const int lane = t & 31, w = t >> 5;

  floatx8 acc[4] = {};

  for (int k0 = 0; k0 < D; k0 += 32) {
    __syncthreads();
#pragma unroll
    for (int j = 0; j < 2; ++j) {       // A tile 64x32, b128 copies
      const int c = t * 2 + j;
      const int r = c >> 2, col = (c & 3) * 8;
      *(uint4*)&As[r * 32 + col] = *(const uint4*)&A[(size_t)(row0 + r) * D + k0 + col];
    }
    stage_bT(&Wh[(size_t)k0 * DQK], DQK, Wl, t);
    __syncthreads();

    const bf16x16 a = load_a_frag(&As[(w * 16) * 32], 32, lane);
#pragma unroll
    for (int j = 0; j < 4; ++j)
      acc[j] = wmma_bf16(a, load_b_fragT(&Wl[(j * 16) * 32], 32, lane), acc[j]);
  }

  // Epilogue: stage C tiles in (now dead) LDS, then 64B coalesced stores.
  __syncthreads();
  __bf16* Cs = (__bf16*)smem_raw;       // 4 waves x 16x64 bf16 = 8KB
  const int n = lane & 15, g = lane >> 4;
#pragma unroll
  for (int j = 0; j < 4; ++j)
#pragma unroll
    for (int r = 0; r < 8; ++r)
      Cs[w * 1024 + (r + 8 * g) * 64 + j * 16 + n] = (__bf16)acc[j][r];
  lds_fence();
  const int lrow = lane >> 1, lcol = (lane & 1) * 32;
  const uint4* src = (const uint4*)&Cs[w * 1024 + lrow * 64 + lcol];
  uint4* dst = (uint4*)&C[(size_t)(row0 + w * 16 + lrow) * DQK + lcol];
#pragma unroll
  for (int u = 0; u < 4; ++u) dst[u] = src[u];
}

// --- Flash attention: per (b, h, 64-query block) ---------------------------

__global__ __launch_bounds__(128) void flash_attn_kernel(
    const __bf16* __restrict__ Qh,           // (B, H, S, 64)
    const __bf16* __restrict__ Kh,           // (B, H, S, 64)
    const __bf16* __restrict__ Vh,           // (B, H, S, 64)
    const unsigned char* __restrict__ mask,  // (B, S, S) bool
    __bf16* __restrict__ Oh) {               // (B, S, H*DV)
  using namespace cfg;
  __shared__ __bf16 Qs[64 * 64];     // 8KB, dead after aQ load -> reused as Cs
  __shared__ __bf16 Ks[32 * 64];     // [key][dqk] row-major == transposed-B
  __shared__ __bf16 Vt[64 * 32];     // [dv][key] transposed
  __shared__ __bf16 Ps[4][16 * 32];  // per-wave P tile (C->A layout bridge)

  const int b  = blockIdx.z;
  const int h  = blockIdx.y;
  const int q0 = blockIdx.x * 64;
  const int t  = threadIdx.x;
  const int lane = t & 31, w = t >> 5;
  const int n = lane & 15, g = lane >> 4;

  const __bf16* Qg = Qh + ((size_t)(b * H + h) * S) * DQK;
  const __bf16* Kg = Kh + ((size_t)(b * H + h) * S) * DQK;
  const __bf16* Vg = Vh + ((size_t)(b * H + h) * S) * DV;

  // Stage the 64x64 Q block once.
#pragma unroll
  for (int j = 0; j < 4; ++j) {
    const int c = t * 4 + j;
    const int r = c >> 3, col = (c & 7) * 8;
    *(uint4*)&Qs[r * 64 + col] = *(const uint4*)&Qg[(size_t)(q0 + r) * DQK + col];
  }
  __syncthreads();

  bf16x16 aQ[2];
#pragma unroll
  for (int kt = 0; kt < 2; ++kt)
    aQ[kt] = load_a_frag(&Qs[(w * 16) * 64 + kt * 32], 64, lane);

  const float NEG_INF = -__builtin_inff();
  float rowMax[8], rowSum[8];
#pragma unroll
  for (int r = 0; r < 8; ++r) { rowMax[r] = NEG_INF; rowSum[r] = 0.f; }
  floatx8 accO[4] = {};

  const unsigned char* mbase = mask + ((size_t)b * S + q0 + w * 16) * S;

  for (int kb = 0; kb < S; kb += 32) {
    __syncthreads();
    // K tile row-major (already transposed-B for QK^T).
#pragma unroll
    for (int j = 0; j < 2; ++j) {
      const int c = t * 2 + j;
      const int r = c >> 3, col = (c & 7) * 8;
      *(uint4*)&Ks[r * 64 + col] = *(const uint4*)&Kg[(size_t)(kb + r) * DQK + col];
    }
    // V tile staged transposed for P·V.
    stage_bT(&Vg[(size_t)kb * DV], DV, Vt, t);
    if (kb + 32 < S) {  // global_prefetch_b8 for the next tile
      __builtin_prefetch(&Kg[(size_t)(kb + 32) * DQK], 0, 3);
      __builtin_prefetch(&Vg[(size_t)(kb + 32) * DV], 0, 3);
    }
    __syncthreads();

    // logits: two 16-key column tiles, contraction over DQK=64 (2 WMMA each).
    floatx8 sfrag[2] = {};
#pragma unroll
    for (int ct = 0; ct < 2; ++ct)
#pragma unroll
      for (int kt = 0; kt < 2; ++kt)
        sfrag[ct] = wmma_bf16(
            aQ[kt], load_b_fragT(&Ks[(ct * 16) * 64 + kt * 32], 64, lane),
            sfrag[ct]);

    // Online softmax per owned row (C layout: row = r + 8g, col = n).
#pragma unroll
    for (int r = 0; r < 8; ++r) {
      const int qrow = r + 8 * g;
      float x0 = sfrag[0][r] * SCALE;
      float x1 = sfrag[1][r] * SCALE;
      if (!mbase[(size_t)qrow * S + kb + n])      x0 = NEG_INF;
      if (!mbase[(size_t)qrow * S + kb + 16 + n]) x1 = NEG_INF;

      float mt = fmaxf(x0, x1);
#pragma unroll
      for (int d = 1; d < 16; d <<= 1) mt = fmaxf(mt, __shfl_xor(mt, d, 32));
      const float newM = fmaxf(rowMax[r], mt);
      const float corr = (rowMax[r] == NEG_INF) ? 0.f : __expf(rowMax[r] - newM);
      const float p0 = (newM == NEG_INF) ? 0.f : __expf(x0 - newM);
      const float p1 = (newM == NEG_INF) ? 0.f : __expf(x1 - newM);
      float ps = p0 + p1;
#pragma unroll
      for (int d = 1; d < 16; d <<= 1) ps += __shfl_xor(ps, d, 32);
      rowSum[r] = rowSum[r] * corr + ps;
      rowMax[r] = newM;
#pragma unroll
      for (int j = 0; j < 4; ++j) accO[j][r] *= corr;

      Ps[w][qrow * 32 + n]      = (__bf16)p0;
      Ps[w][qrow * 32 + 16 + n] = (__bf16)p1;
    }
    lds_fence();

    const bf16x16 aP = load_a_frag(&Ps[w][0], 32, lane);
#pragma unroll
    for (int j = 0; j < 4; ++j)
      accO[j] = wmma_bf16(aP, load_b_fragT(&Vt[(j * 16) * 32], 32, lane), accO[j]);
  }

  // Epilogue: normalize, stage in (dead) Qs, 64B coalesced stores.
  __syncthreads();
  __bf16* Cs = Qs;                         // 4 waves x 16x64 bf16 = 8KB
#pragma unroll
  for (int r = 0; r < 8; ++r) {
    const float inv = 1.0f / rowSum[r];
#pragma unroll
    for (int j = 0; j < 4; ++j)
      Cs[w * 1024 + (r + 8 * g) * 64 + j * 16 + n] = (__bf16)(accO[j][r] * inv);
  }
  lds_fence();
  const int lrow = lane >> 1, lcol = (lane & 1) * 32;
  const uint4* src = (const uint4*)&Cs[w * 1024 + lrow * 64 + lcol];
  uint4* dst = (uint4*)&Oh[((size_t)b * S + q0 + w * 16 + lrow) * HDV + h * DV + lcol];
#pragma unroll
  for (int u = 0; u < 4; ++u) dst[u] = src[u];
}

// --- Output projection: (B*S x 1024) @ (1024 x 1024) -> f32 ----------------

__global__ __launch_bounds__(128) void out_gemm_kernel(
    const __bf16* __restrict__ A,    // (B*S, HDV)
    const __bf16* __restrict__ W,    // (HDV, DOUT)
    float* __restrict__ C) {         // (B*S, DOUT) f32
  using namespace cfg;
  __shared__ __align__(16) char smem_raw[16384];
  __bf16* As = (__bf16*)smem_raw;            // 64x32 row-major (4KB)
  __bf16* Wl = (__bf16*)(smem_raw + 4096);   // 64(n) x 32(k) T (4KB)

  const int row0 = blockIdx.x * 64;
  const int n0   = blockIdx.y * 64;
  const int t    = threadIdx.x;
  const int lane = t & 31, w = t >> 5;

  floatx8 acc[4] = {};

  for (int k0 = 0; k0 < HDV; k0 += 32) {
    __syncthreads();
#pragma unroll
    for (int j = 0; j < 2; ++j) {
      const int c = t * 2 + j;
      const int r = c >> 2, col = (c & 3) * 8;
      *(uint4*)&As[r * 32 + col] = *(const uint4*)&A[(size_t)(row0 + r) * HDV + k0 + col];
    }
    stage_bT(&W[(size_t)k0 * DOUT + n0], DOUT, Wl, t);
    __syncthreads();

    const bf16x16 a = load_a_frag(&As[(w * 16) * 32], 32, lane);
#pragma unroll
    for (int j = 0; j < 4; ++j)
      acc[j] = wmma_bf16(a, load_b_fragT(&Wl[(j * 16) * 32], 32, lane), acc[j]);
  }

  // Epilogue through LDS (f32 tiles, 4 waves x 16x64 f32 = 16KB).
  __syncthreads();
  float* Cs = (float*)smem_raw;
  const int n = lane & 15, g = lane >> 4;
#pragma unroll
  for (int j = 0; j < 4; ++j)
#pragma unroll
    for (int r = 0; r < 8; ++r)
      Cs[w * 1024 + (r + 8 * g) * 64 + j * 16 + n] = acc[j][r];
  lds_fence();
  const int lrow = lane >> 1, lcol = (lane & 1) * 32;
  const uint4* src = (const uint4*)&Cs[w * 1024 + lrow * 64 + lcol];
  uint4* dst = (uint4*)&C[(size_t)(row0 + w * 16 + lrow) * DOUT + n0 + lcol];
#pragma unroll
  for (int u = 0; u < 8; ++u) dst[u] = src[u];
}

// ---------------------------------------------------------------------------

extern "C" void kernel_launch(void* const* d_in, const int* in_sizes, int n_in,
                              void* d_out, int out_size, void* d_ws, size_t ws_size,
                              hipStream_t stream) {
  using namespace cfg;
  (void)in_sizes; (void)n_in; (void)out_size; (void)ws_size;

  const float* q_in  = (const float*)d_in[0];
  const float* k_in  = (const float*)d_in[1];
  const float* v_in  = (const float*)d_in[2];
  const unsigned char* mask = (const unsigned char*)d_in[3];
  const float* Wq = (const float*)d_in[4];
  const float* Wk = (const float*)d_in[5];
  const float* Wv = (const float*)d_in[6];
  const float* Wo = (const float*)d_in[7];
  float* out = (float*)d_out;

  // Workspace carve-up (bf16, 256B-aligned slabs), ~67 MB total.
  char* ws = (char*)d_ws;
  size_t off = 0;
  auto alloc_bf = [&](size_t elems) {
    __bf16* p = (__bf16*)(ws + off);
    off += ((elems * sizeof(__bf16) + 255) / 256) * 256;
    return p;
  };
  __bf16* Xq  = alloc_bf((size_t)B * S * D);
  __bf16* Xk  = alloc_bf((size_t)B * S * D);
  __bf16* Xv  = alloc_bf((size_t)B * S * D);
  __bf16* Wqb = alloc_bf((size_t)H * D * DQK);
  __bf16* Wkb = alloc_bf((size_t)H * D * DQK);
  __bf16* Wvb = alloc_bf((size_t)H * D * DV);
  __bf16* Wob = alloc_bf((size_t)HDV * DOUT);
  __bf16* Qh  = alloc_bf((size_t)B * H * S * DQK);
  __bf16* Kh  = alloc_bf((size_t)B * H * S * DQK);
  __bf16* Vh  = alloc_bf((size_t)B * H * S * DV);
  __bf16* Oh  = alloc_bf((size_t)B * S * HDV);

  auto cvt = [&](const float* src, __bf16* dst, int n) {
    cvt_f32_bf16_kernel<<<(n + 255) / 256, 256, 0, stream>>>(src, dst, n);
  };
  cvt(q_in, Xq, B * S * D);
  cvt(k_in, Xk, B * S * D);
  cvt(v_in, Xv, B * S * D);
  cvt(Wq, Wqb, H * D * DQK);
  cvt(Wk, Wkb, H * D * DQK);
  cvt(Wv, Wvb, H * D * DV);
  cvt(Wo, Wob, HDV * DOUT);

  // Per-head projections: batched over (b,h).
  dim3 pgrid(S / 64, 1, B * H);
  proj_gemm_kernel<<<pgrid, 128, 0, stream>>>(Xq, Wqb, Qh);
  proj_gemm_kernel<<<pgrid, 128, 0, stream>>>(Xk, Wkb, Kh);
  proj_gemm_kernel<<<pgrid, 128, 0, stream>>>(Xv, Wvb, Vh);

  // Fused softmax attention.
  dim3 agrid(S / 64, H, B);
  flash_attn_kernel<<<agrid, 128, 0, stream>>>(Qh, Kh, Vh, mask, Oh);

  // Output projection.
  dim3 ogrid((B * S) / 64, DOUT / 64, 1);
  out_gemm_kernel<<<ogrid, 128, 0, stream>>>(Oh, Wob, out);
}
